// IterativeConvLRU_80771154969252
// MI455X (gfx1250) — matrix operand order repaired and checked by reading, not compiled, not run on hardware
//
#include <hip/hip_runtime.h>
#include <math.h>

// ---------------- problem constants ----------------
#define NBk 2
#define Bb  4
#define Ll  16
#define CIc 1
#define Ee  64
#define CHh 128
#define Ss  32
#define OFf 4
#define NS  (Bb*Ll)     // 64 samples
#define PIX (Ss*Ss)     // 1024 pixels

typedef __attribute__((ext_vector_type(16))) _Float16 v16h;
typedef __attribute__((ext_vector_type(8)))  float    v8f;
typedef __attribute__((ext_vector_type(4)))  unsigned int u32x4;
typedef __attribute__((ext_vector_type(8)))  int          i32x8;
typedef __attribute__((ext_vector_type(4)))  int          i32x4;

#define HAVE_TDM (__has_builtin(__builtin_amdgcn_tensor_load_to_lds) && \
                  __has_builtin(__builtin_amdgcn_s_wait_tensorcnt))

// ==========================================================================
// Implicit-GEMM 3x3 SAME conv on 32x32 maps, f32 in/out, f16 WMMA, f32 acc.
// Block (256 thr = 8 waves) owns: one sample, one 32-wide cout slab, 8 pixel
// tiles. Weight slab (32 x CIN*9 f32) is DMA'd to LDS by the Tensor Data
// Mover (wave 0 issues TENSOR_LOAD_TO_LDS, waits TENSORcnt), then every wave
// computes TWO 16x16 WMMA accumulators sharing one A fragment.
//   A (16x32 MxK): lane m=lane&15 holds row M=m; half=lane>>4 selects K subset
//     {half*16..+15}: contiguous k order, same as B.
//   B (32x16 KxN): lane holds N=lane&15, K=(lane>>4)*16 + j  (from LDS).
//   C (16x16 f32): acc[r] = C[M = r + half*8][N = lane&15].
// ==========================================================================
template<int CIN, bool GELU>
__global__ __launch_bounds__(256)
void conv3_wmma(const float* __restrict__ x, const float* __restrict__ w,
                const float* __restrict__ bias, float* __restrict__ y, int Cout) {
  constexpr int KTOT   = CIN * 9;
  constexpr int KSTEPS = (KTOT + 31) >> 5;
  extern __shared__ float wlds[];              // [32][KTOT] f32, row-contiguous

  const int cgroups = (Cout + 31) >> 5;
  int bx = blockIdx.x;
  const int mg = bx & 7;  bx >>= 3;
  const int cg = bx % cgroups;
  const int nt = bx / cgroups;
  const int cobase = cg << 5;

  // ---- stage the 32-row weight slab into LDS ----
#if HAVE_TDM
  if (threadIdx.x < 32) {                      // wave 0 issues the DMA
    const unsigned long long gaddr =
        (unsigned long long)(size_t)(w + (size_t)cobase * KTOT);
    const unsigned int laddr = (unsigned int)(size_t)&wlds[0]; // LDS = addr[31:0]
    int rows = Cout - cobase; if (rows > 32) rows = 32;        // OOB rows -> 0
    u32x4 g0; i32x8 g1; i32x4 gz4; i32x8 gz8;
    g0[0] = 1u;                                                // count=1 (valid)
    g0[1] = laddr;                                             // lds_addr
    g0[2] = (unsigned int)gaddr;                               // global_addr lo
    g0[3] = ((unsigned int)(gaddr >> 32) & 0x01FFFFFFu)
            | 0x80000000u;                                     // type=2 (image)
    g1[0] = 0x20000;                            // data_size=2 (4B), no pad
    g1[1] = (int)((KTOT & 0xFFFF) << 16);       // tensor_dim0[15:0]
    g1[2] = (int)((rows & 0xFFFF) << 16);       // dim0 hi=0 | tensor_dim1[15:0]
    g1[3] = (int)((KTOT & 0xFFFF) << 16);       // dim1 hi=0 | tile_dim0
    g1[4] = 32;                                 // tile_dim1=32, tile_dim2=0
    g1[5] = KTOT;                               // tensor_dim0_stride lo32
    g1[6] = 0; g1[7] = 0;
    gz4[0] = 0; gz4[1] = 0; gz4[2] = 0; gz4[3] = 0;
    gz8[0] = 0; gz8[1] = 0; gz8[2] = 0; gz8[3] = 0;
    gz8[4] = 0; gz8[5] = 0; gz8[6] = 0; gz8[7] = 0;
    // 6-arg (clang-23 / therock) form: groups 2/3 + trailing group zeroed
    __builtin_amdgcn_tensor_load_to_lds(g0, g1, gz4, gz4, gz8, 0);
    __builtin_amdgcn_s_wait_tensorcnt(0);
  }
#else
  for (int idx = threadIdx.x; idx < 32 * KTOT; idx += 256) {
    const int rowi = idx / KTOT, k = idx - rowi * KTOT;
    const int co = cobase + rowi;
    wlds[idx] = (co < Cout) ? w[(size_t)co * KTOT + k] : 0.0f;
  }
#endif
  __syncthreads();

  const int waveId = threadIdx.x >> 5;
  const int lane   = threadIdx.x & 31;
  const int mt     = mg * 8 + waveId;           // pixel tile 0..63
  const int row  = mt >> 1;
  const int col0 = (mt & 1) << 4;
  const int m    = lane & 15;
  const int half = lane >> 4;
  const int px   = col0 + m;                    // this lane's A-row pixel col
  const float* __restrict__ xn = x + (size_t)nt * CIN * PIX;

  v8f acc0 = {}, acc1 = {};
  for (int ks = 0; ks < KSTEPS; ++ks) {
    if (ks + 1 < KSTEPS) {  // warm next input-channel slab (global_prefetch_b8)
      const int cnx = ((ks + 1) * 32) / 9;
      __builtin_prefetch(&xn[(cnx * 32 + row) * 32 + px], 0, 0);
    }
    v16h a, b0, b1;
#pragma unroll
    for (int j = 0; j < 16; ++j) {
      const int kl = half * 16 + j;
      const int K  = ks * 32 + kl;
      float v = 0.0f;
      if (K < KTOT) {
        const int cin = K / 9, r9 = K - cin * 9;
        const int iy = row + r9 / 3 - 1;
        const int ix = px  + (r9 % 3) - 1;
        if ((unsigned)iy < 32u && (unsigned)ix < 32u)
          v = xn[(cin * 32 + iy) * 32 + ix];
      }
      a[j] = (_Float16)v;
    }
#pragma unroll
    for (int j = 0; j < 16; ++j) {              // B fragments from LDS
      const int K = ks * 32 + half * 16 + j;
      float v0 = 0.0f, v1 = 0.0f;
      if (K < KTOT) {
        v0 = wlds[m * KTOT + K];
        v1 = wlds[(16 + m) * KTOT + K];
      }
      b0[j] = (_Float16)v0;
      b1[j] = (_Float16)v1;
    }
    acc0 = __builtin_amdgcn_wmma_f32_16x16x32_f16(false, a, false, b0,
                                                  (short)0, acc0, false, false);
    acc1 = __builtin_amdgcn_wmma_f32_16x16x32_f16(false, a, false, b1,
                                                  (short)0, acc1, false, false);
  }
#pragma unroll
  for (int r = 0; r < 8; ++r) {
    const int Mi  = r + half * 8;
    const int co0 = cobase + m;
    const int co1 = cobase + 16 + m;
    const size_t pofs = row * 32 + col0 + Mi;
    if (co0 < Cout) {
      float val = acc0[r] + bias[co0];
      if (GELU) val = 0.5f * val * (1.0f + erff(val * 0.70710678118f));
      y[((size_t)nt * Cout + co0) * PIX + pofs] = val;
    }
    if (co1 < Cout) {
      float val = acc1[r] + bias[co1];
      if (GELU) val = 0.5f * val * (1.0f + erff(val * 0.70710678118f));
      y[((size_t)nt * Cout + co1) * PIX + pofs] = val;
    }
  }
}

// ==========================================================================
// LRU projection: gamma*eye masking means only diagonal pixels survive.
// hdiag[n,c,i] = gamma[c,i] * (sum_e x[n,e,i,i]*Wp[c,e] + bp[c])  (complex)
// GEMM: M = NS*32 = 2048 rows (n,i), K = E = 64, N = CH = 128; dual acc re/im.
// ==========================================================================
__global__ __launch_bounds__(256)
void lru_project_wmma(const float* __restrict__ h,
                      const float* __restrict__ Wpre, const float* __restrict__ Wpim,
                      const float* __restrict__ bpre, const float* __restrict__ bpim,
                      const float* __restrict__ plog,   // (3*CH, 32) for this block
                      float* __restrict__ hre, float* __restrict__ him) {
  const int wave = (int)((blockIdx.x * blockDim.x + threadIdx.x) >> 5);
  const int lane = threadIdx.x & 31;
  const int ctiles = CHh / 16;                 // 8
  const int mt = wave / ctiles, ct = wave % ctiles;
  const int m = lane & 15, half = lane >> 4;
  const int mrow = mt * 16 + m;
  const int n = mrow >> 5, di = mrow & 31;

  v8f accR = {}, accI = {};
  for (int ks = 0; ks < Ee / 32; ++ks) {       // exactly 2 steps
    v16h a, br, bi;
#pragma unroll
    for (int j = 0; j < 16; ++j) {
      const int kl = half * 16 + j;
      const int e  = ks * 32 + kl;
      a[j] = (_Float16)h[((size_t)n * Ee + e) * PIX + di * 33];   // diag (i,i)
    }
#pragma unroll
    for (int j = 0; j < 16; ++j) {
      const int e = ks * 32 + half * 16 + j;
      const int c = ct * 16 + m;
      br[j] = (_Float16)Wpre[c * Ee + e];
      bi[j] = (_Float16)Wpim[c * Ee + e];
    }
    accR = __builtin_amdgcn_wmma_f32_16x16x32_f16(false, a, false, br,
                                                  (short)0, accR, false, false);
    accI = __builtin_amdgcn_wmma_f32_16x16x32_f16(false, a, false, bi,
                                                  (short)0, accI, false, false);
  }
#pragma unroll
  for (int r = 0; r < 8; ++r) {
    const int mr = mt * 16 + r + half * 8;
    const int nn = mr >> 5, ii = mr & 31;
    const int c  = ct * 16 + m;
    const float g = __expf(plog[(2 * CHh + c) * 32 + ii]);
    hre[((size_t)nn * CHh + c) * 32 + ii] = g * (accR[r] + bpre[c]);
    him[((size_t)nn * CHh + c) * 32 + ii] = g * (accI[r] + bpim[c]);
  }
}

// ==========================================================================
// Diagonal LRU scan: 16384 independent length-16 complex recurrences.
// ==========================================================================
__global__ void lru_scan_k(float* __restrict__ hre, float* __restrict__ him,
                           const float* __restrict__ plog) {
  const int idx = blockIdx.x * blockDim.x + threadIdx.x;
  if (idx >= Bb * CHh * Ss) return;
  const int i = idx & 31;
  const int c = (idx >> 5) & (CHh - 1);
  const int b = idx >> 12;
  const float nu = __expf(plog[c * 32 + i]);
  const float th = __expf(plog[(CHh + c) * 32 + i]);
  const float rr = __expf(-nu);
  const float lr = rr * __cosf(th), li = rr * __sinf(th);
  float pr = 0.0f, pi = 0.0f;
  for (int t = 0; t < Ll; ++t) {
    const size_t o = ((size_t)(b * Ll + t) * CHh + c) * 32 + i;
    const float nr = hre[o] + lr * pr - li * pi;
    const float ni = him[o] + lr * pi + li * pr;
    hre[o] = nr; him[o] = ni;
    pr = nr; pi = ni;
  }
}

// ==========================================================================
// Output projection (real part only survives):
// yd[n,e,i] = sum_c hre[n,c,i]*Wc_re[e,c] - him[n,c,i]*Wc_im[e,c]
// ==========================================================================
__global__ __launch_bounds__(256)
void lru_out_wmma(const float* __restrict__ hre, const float* __restrict__ him,
                  const float* __restrict__ Wcre, const float* __restrict__ Wcim,
                  float* __restrict__ yd) {
  const int wave = (int)((blockIdx.x * blockDim.x + threadIdx.x) >> 5);
  const int lane = threadIdx.x & 31;
  const int etiles = Ee / 16;                  // 4
  const int mt = wave / etiles, et = wave % etiles;
  const int m = lane & 15, half = lane >> 4;
  const int mrow = mt * 16 + m;
  const int n = mrow >> 5, di = mrow & 31;

  v8f acc = {};
  for (int part = 0; part < 2; ++part) {
    const float* __restrict__ A  = part ? him  : hre;
    const float* __restrict__ Bw = part ? Wcim : Wcre;
    const float sgn = part ? -1.0f : 1.0f;
    for (int ks = 0; ks < CHh / 32; ++ks) {    // 4 steps per pass
      v16h a, b;
#pragma unroll
      for (int j = 0; j < 16; ++j) {
        const int kl = half * 16 + j;
        const int c  = ks * 32 + kl;
        a[j] = (_Float16)(sgn * A[((size_t)n * CHh + c) * 32 + di]);
      }
#pragma unroll
      for (int j = 0; j < 16; ++j) {
        const int c = ks * 32 + half * 16 + j;
        const int e = et * 16 + m;
        b[j] = (_Float16)Bw[e * CHh + c];
      }
      acc = __builtin_amdgcn_wmma_f32_16x16x32_f16(false, a, false, b,
                                                   (short)0, acc, false, false);
    }
  }
#pragma unroll
  for (int r = 0; r < 8; ++r) {
    const int mr = mt * 16 + r + half * 8;
    const int nn = mr >> 5, ii = mr & 31;
    const int e  = et * 16 + m;
    yd[((size_t)nn * Ee + e) * 32 + ii] = acc[r];
  }
}

// ==========================================================================
// ifft2 of (Re(bc[e]) + diag(yd)).real:
//   out[k1,k2] = (1/1024) sum_i yd[i] cos(2*pi*i*(k1+k2)/32)  + bc_re[e]@(0,0)
// ==========================================================================
__global__ void ifft_diag_k(const float* __restrict__ yd,
                            const float* __restrict__ bcre,
                            float* __restrict__ y) {
  const int n = blockIdx.x >> 6;
  const int e = blockIdx.x & 63;
  const int t = threadIdx.x;                   // frequency index 0..31
  __shared__ float g[32];
  float s = 0.0f;
  const float* __restrict__ d = yd + ((size_t)n * Ee + e) * 32;
#pragma unroll 8
  for (int i = 0; i < 32; ++i)
    s += d[i] * __cosf(0.19634954084936207f * (float)((i * t) & 31)); // 2pi/32
  g[t] = s * (1.0f / 1024.0f);
  __syncthreads();
  const float bc = bcre[e];
  float* __restrict__ yp = y + ((size_t)n * Ee + e) * PIX;
  for (int k1 = 0; k1 < 32; ++k1) {
    float v = g[(k1 + t) & 31];
    if (k1 == 0 && t == 0) v += bc;
    yp[k1 * 32 + t] = v;
  }
}

// ==========================================================================
// LayerNorm over (E,32,32) per sample + affine + residual (in-place on hres).
// ==========================================================================
__global__ __launch_bounds__(1024)
void ln_residual_k(const float* __restrict__ yin,
                   const float* __restrict__ lnw, const float* __restrict__ lnb,
                   float* __restrict__ hres) {
  const int n = blockIdx.x;
  const float* __restrict__ yp = yin + (size_t)n * Ee * PIX;
  float* __restrict__ hp = hres + (size_t)n * Ee * PIX;
  __shared__ float s1[32], s2[32];
  float sum = 0.0f, sq = 0.0f;
  for (int i = threadIdx.x; i < Ee * PIX; i += 1024) {
    const float v = yp[i];
    sum += v; sq += v * v;
  }
#pragma unroll
  for (int off = 16; off; off >>= 1) {
    sum += __shfl_down(sum, off, 32);
    sq  += __shfl_down(sq,  off, 32);
  }
  if ((threadIdx.x & 31) == 0) { s1[threadIdx.x >> 5] = sum; s2[threadIdx.x >> 5] = sq; }
  __syncthreads();
  if (threadIdx.x < 32) {
    sum = s1[threadIdx.x]; sq = s2[threadIdx.x];
#pragma unroll
    for (int off = 16; off; off >>= 1) {
      sum += __shfl_down(sum, off, 32);
      sq  += __shfl_down(sq,  off, 32);
    }
    if (threadIdx.x == 0) { s1[0] = sum; s2[0] = sq; }
  }
  __syncthreads();
  const float inv  = 1.0f / (float)(Ee * PIX);
  const float mean = s1[0] * inv;
  const float var  = s2[0] * inv - mean * mean;
  const float rstd = rsqrtf(var + 1e-5f);
  for (int i = threadIdx.x; i < Ee * PIX; i += 1024)
    hp[i] = (yp[i] - mean) * rstd * lnw[i] + lnb[i] + hp[i];
}

// ==========================================================================
// Window shift + forecast-frame extraction.
// ==========================================================================
__global__ void shift_out_k(const float* __restrict__ cur,
                            const float* __restrict__ decb,
                            float* __restrict__ nxt, float* __restrict__ out,
                            int of) {
  const int idx = blockIdx.x * blockDim.x + threadIdx.x;
  if (idx >= Bb * Ll * PIX) return;
  const int p = idx & 1023;
  const int t = (idx >> 10) & 15;
  const int b = idx >> 14;
  const float v = (t < Ll - 1) ? cur[(size_t)(b * Ll + t + 1) * PIX + p]
                               : decb[(size_t)(b * Ll + Ll - 1) * PIX + p];
  nxt[(size_t)(b * Ll + t) * PIX + p] = v;
  if (t == Ll - 1) out[(size_t)(b * OFf + of) * PIX + p] = v;
}

// ==========================================================================
extern "C" void kernel_launch(void* const* d_in, const int* in_sizes, int n_in,
                              void* d_out, int out_size, void* d_ws, size_t ws_size,
                              hipStream_t stream) {
  (void)in_sizes; (void)n_in; (void)out_size; (void)ws_size;
  const float* x          = (const float*)d_in[0];
  const float* emb_w      = (const float*)d_in[1];
  const float* emb_b      = (const float*)d_in[2];
  const float* dec_w      = (const float*)d_in[3];
  const float* dec_b      = (const float*)d_in[4];
  const float* params_log = (const float*)d_in[5];
  const float* Wp_re      = (const float*)d_in[6];
  const float* Wp_im      = (const float*)d_in[7];
  const float* bp_re      = (const float*)d_in[8];
  const float* bp_im      = (const float*)d_in[9];
  const float* Wc_re      = (const float*)d_in[10];
  const float* Wc_im      = (const float*)d_in[11];
  const float* bc_re      = (const float*)d_in[12];
  // d_in[13] = bc_im: unused (only the real part of y survives .real)
  const float* ln1_w      = (const float*)d_in[14];
  const float* ln1_b      = (const float*)d_in[15];
  const float* w1         = (const float*)d_in[16];
  const float* b1         = (const float*)d_in[17];
  const float* w2         = (const float*)d_in[18];
  const float* b2         = (const float*)d_in[19];
  const float* ln2_w      = (const float*)d_in[20];
  const float* ln2_b      = (const float*)d_in[21];
  float* out = (float*)d_out;

  // workspace layout (floats); total ~17.6M floats (~70 MB), L2-resident
  float* ws = (float*)d_ws;
  size_t o = 0;
  float* h    = ws + o; o += (size_t)NS * Ee  * PIX;   // activations
  float* tmp  = ws + o; o += (size_t)NS * Ee  * PIX;   // pre-LN buffer
  float* mid  = ws + o; o += (size_t)NS * CHh * PIX;   // FFN hidden
  float* hre  = ws + o; o += (size_t)NS * CHh * Ss;
  float* him  = ws + o; o += (size_t)NS * CHh * Ss;
  float* yd   = ws + o; o += (size_t)NS * Ee  * Ss;
  float* decb = ws + o; o += (size_t)NS * CIc * PIX;
  float* bufA = ws + o; o += (size_t)Bb * Ll * PIX;
  float* bufB = ws + o; o += (size_t)Bb * Ll * PIX;

  // dynamic-LDS sizes for the weight slabs: 32 rows x CIN*9 f32
  const unsigned smem_emb = 32 * (CIc * 9) * 4;   // ~1.2 KB
  const unsigned smem_c1  = 32 * (Ee  * 9) * 4;   // 72 KB
  const unsigned smem_c2  = 32 * (CHh * 9) * 4;   // 144 KB (<=320 KB/WGP LDS)
  const unsigned smem_dec = 32 * (Ee  * 9) * 4;   // 72 KB

  const float* cur = x;
  for (int of = 0; of < OFf; ++of) {
    // --- embedding conv: CI(1) -> E(64); grid = NS * (64/32) * 8
    conv3_wmma<CIc, false><<<NS * 2 * 8, 256, smem_emb, stream>>>(
        cur, emb_w, emb_b, h, Ee);

    for (int k = 0; k < NBk; ++k) {
      const float* plogk = params_log + (size_t)k * 3 * CHh * Ss;
      // --- LRU layer (diagonal-collapsed) ---
      lru_project_wmma<<<(NS * Ss / 16) * (CHh / 16) / 8, 256, 0, stream>>>(
          h, Wp_re + (size_t)k * CHh * Ee, Wp_im + (size_t)k * CHh * Ee,
          bp_re + (size_t)k * CHh, bp_im + (size_t)k * CHh, plogk, hre, him);
      lru_scan_k<<<(Bb * CHh * Ss + 255) / 256, 256, 0, stream>>>(hre, him, plogk);
      lru_out_wmma<<<(NS * Ss / 16) * (Ee / 16) / 8, 256, 0, stream>>>(
          hre, him, Wc_re + (size_t)k * Ee * CHh, Wc_im + (size_t)k * Ee * CHh, yd);
      ifft_diag_k<<<NS * Ee, 32, 0, stream>>>(yd, bc_re + (size_t)k * Ee, tmp);
      ln_residual_k<<<NS, 1024, 0, stream>>>(
          tmp, ln1_w + (size_t)k * Ee * PIX, ln1_b + (size_t)k * Ee * PIX, h);
      // --- FFN: conv3(E->CH)+GELU, conv3(CH->E), LN+residual ---
      conv3_wmma<Ee, true><<<NS * 4 * 8, 256, smem_c1, stream>>>(
          h, w1 + (size_t)k * CHh * Ee * 9, b1 + (size_t)k * CHh, mid, CHh);
      conv3_wmma<CHh, false><<<NS * 2 * 8, 256, smem_c2, stream>>>(
          mid, w2 + (size_t)k * Ee * CHh * 9, b2 + (size_t)k * Ee, tmp, Ee);
      ln_residual_k<<<NS, 1024, 0, stream>>>(
          tmp, ln2_w + (size_t)k * Ee * PIX, ln2_b + (size_t)k * Ee * PIX, h);
    }

    // --- decoder conv: E -> CI(1); one padded 32-slab, TDM zero-fills rows
    conv3_wmma<Ee, false><<<NS * 1 * 8, 256, smem_dec, stream>>>(
        h, dec_w, dec_b, decb, CIc);

    // --- shift window + emit forecast frame (double-buffered, capture-safe)
    float* nxt = (of & 1) ? bufB : bufA;
    shift_out_k<<<(Bb * Ll * PIX + 255) / 256, 256, 0, stream>>>(
        cur, decb, nxt, out, of);
    cur = nxt;
  }
}
